// ChaosKernel_53214644798130
// MI455X (gfx1250) — matrix operand order repaired and checked by reading, not compiled, not run on hardware
//
#include <hip/hip_runtime.h>
#include <math.h>

// ---------------------------------------------------------------------------
// Model dims (from reference)
#define BATCH   8
#define SEQL    512
#define DMODEL  1024
#define NHEADS  16
#define HDIM    64
#define NLAYERS 4
#define VOCAB   32000
#define NBASIN  64
#define MTOK    (BATCH*SEQL)   // 4096 tokens
#define FFDIM   (4*DMODEL)     // 4096

// ---------------------------------------------------------------------------
// WMMA types (CDNA5 wave32, V_WMMA_F32_16X16X32_BF16)
typedef __attribute__((ext_vector_type(16))) __bf16         v16bf;
typedef __attribute__((ext_vector_type(16))) unsigned short v16us;
typedef __attribute__((ext_vector_type(8)))  float          v8f;

__device__ inline unsigned short f2bf(float f) {
  unsigned int u = __builtin_bit_cast(unsigned int, f);
  unsigned int r = u + 0x7FFFu + ((u >> 16) & 1u);   // round-to-nearest-even
  return (unsigned short)(r >> 16);
}
__device__ inline float sigf(float x)  { return 1.0f / (1.0f + expf(-x)); }
__device__ inline float geluf(float x) { return 0.5f * x * (1.0f + erff(x * 0.70710678118654752f)); }

// Fragment layouts (bf16, 16x16x32 WMMA, wave32):
//  A-frag (16x32): lane l -> m = l&15 ; elem e -> k = (e<8)? 8*hi+e : 16+8*hi+(e-8), hi=l>>4
//  B-frag (32x16): lane l -> n = l&15 ; elem e -> k = (e<8)? 8*hi+e : 16+8*hi+(e-8) (same as A)
// Tiled storage: one fragment = 512 contiguous ushorts = [lane][e].
//  A-tiled: tile (mt,kt) at ((mt*(K/32)+kt)*512)
//  B-tiled: tile (kt,nt) at ((kt*(N/16)+nt)*512)

// ---------------------------------------------------------------------------
// f32 [M,K] (row stride rs) -> A-fragment-tiled bf16. Slices for per-(b,h) views.
__global__ __launch_bounds__(256)
void k_cvtA(const float* __restrict__ src, unsigned short* __restrict__ dst,
            int M, int K, int rs, int innerN, long srcInner, long srcOuter)
{
  const int g = blockIdx.x * 256 + threadIdx.x;
  const int KT = K >> 5;
  const int perSlice = (M >> 4) * KT * 32;
  const int sl = g / perSlice;
  int r = g - sl * perSlice;
  const int l = r & 31; r >>= 5;
  const int kt = r % KT, mt = r / KT;
  const int hi = l >> 4, ln = l & 15;
  const float* s = src + (long)(sl / innerN) * srcOuter + (long)(sl % innerN) * srcInner
                       + (long)(mt * 16 + ln) * rs + kt * 32;
  unsigned short* d = dst + (long)sl * ((long)M * K) + ((long)(mt * KT + kt) * 512 + l * 16);
  #pragma unroll
  for (int e = 0; e < 16; ++e) {
    const int k = (e < 8) ? (8 * hi + e) : (16 + 8 * hi + (e - 8));
    d[e] = f2bf(s[k]);
  }
}
// f32 K x N (arbitrary strides) -> B-fragment-tiled bf16. Slices as above.
__global__ __launch_bounds__(256)
void k_cvtB(const float* __restrict__ src, unsigned short* __restrict__ dst,
            int K, int N, long kStride, long nStride,
            int innerN, long srcInner, long srcOuter)
{
  const int g = blockIdx.x * 256 + threadIdx.x;
  const int NT = N >> 4;
  const int perSlice = (K >> 5) * NT * 32;
  const int sl = g / perSlice;
  int r = g - sl * perSlice;
  const int l = r & 31; r >>= 5;
  const int nt = r % NT, kt = r / NT;
  const int hi = l >> 4, ln = l & 15;
  const float* s = src + (long)(sl / innerN) * srcOuter + (long)(sl % innerN) * srcInner
                       + (long)(kt * 32) * kStride + (long)(nt * 16 + ln) * nStride;
  unsigned short* d = dst + (long)sl * ((long)K * N) + ((long)(kt * NT + nt) * 512 + l * 16);
  #pragma unroll
  for (int e = 0; e < 16; ++e) d[e] = f2bf(s[(long)((e < 8) ? (8 * hi + e) : (16 + 8 * hi + (e - 8))) * kStride]);
}

// ---------------------------------------------------------------------------
// WMMA GEMM on pre-tiled bf16: C[M,N] = epi(At @ Bt + bias [+R]).
// Block 128x128, 8 waves (32x64 each, 2x4 WMMA tiles). Double-buffered LDS fed
// by CDNA5 async loads (global_load_async_to_lds_b128 + s_wait_asynccnt).
// EPI: 0=none 1=gelu 2=sigmoid 3=residual-add
template <int EPI>
__global__ __launch_bounds__(256)
void k_gemm(const unsigned short* __restrict__ At, const unsigned short* __restrict__ Bt,
            const float* __restrict__ bias, const float* R,
            float* C, int M, int N, int K)
{
  __shared__ __attribute__((aligned(32))) unsigned short lA[2][8 * 512];
  __shared__ __attribute__((aligned(32))) unsigned short lB[2][8 * 512];

  const int tid = threadIdx.x;
  const int l   = tid & 31, ln = l & 15, hi = l >> 4;
  const int sub = tid >> 5;                    // staging sub-tile == wave id
  const int waveM = sub & 3, waveN = sub >> 2; // 4x2 wave grid
  const int KT = K >> 5, NT = N >> 4;
  const int mtBase = blockIdx.y << 3;          // rowBase/16
  const int ntBase = blockIdx.x << 3;          // colBase/16
  const unsigned long long aBase = (unsigned long long)At;
  const unsigned long long bBase = (unsigned long long)Bt;

  auto stage = [&](int kt, int buf) { // 4 async b128 loads per thread (32B A + 32B B)
    const unsigned aOff = (unsigned)((((mtBase + sub) * KT + kt) * 512 + l * 16) * 2);
    const unsigned bOff = (unsigned)(((kt * NT + ntBase + sub) * 512 + l * 16) * 2);
    const unsigned aLds = (unsigned)(size_t)&lA[buf][sub * 512 + l * 16];
    const unsigned bLds = (unsigned)(size_t)&lB[buf][sub * 512 + l * 16];
    asm volatile("global_load_async_to_lds_b128 %0, %1, %2 offset:0"
                 :: "v"(aLds), "v"(aOff), "s"(aBase) : "memory");
    asm volatile("global_load_async_to_lds_b128 %0, %1, %2 offset:16"
                 :: "v"(aLds), "v"(aOff), "s"(aBase) : "memory");
    asm volatile("global_load_async_to_lds_b128 %0, %1, %2 offset:0"
                 :: "v"(bLds), "v"(bOff), "s"(bBase) : "memory");
    asm volatile("global_load_async_to_lds_b128 %0, %1, %2 offset:16"
                 :: "v"(bLds), "v"(bOff), "s"(bBase) : "memory");
  };

  const v8f vz = {0.f, 0.f, 0.f, 0.f, 0.f, 0.f, 0.f, 0.f};
  v8f acc[2][4];
  #pragma unroll
  for (int mi = 0; mi < 2; ++mi)
    #pragma unroll
    for (int ni = 0; ni < 4; ++ni) acc[mi][ni] = vz;

  stage(0, 0);
  for (int kt = 0; kt < KT; ++kt) {
    const int cur = kt & 1;
    if (kt + 1 < KT) {
      stage(kt + 1, cur ^ 1);                        // prefetch next tile
      asm volatile("s_wait_asynccnt 0x4" ::: "memory"); // current tile's 4 done
    } else {
      asm volatile("s_wait_asynccnt 0x0" ::: "memory");
    }
    __syncthreads();

    v16bf aF[2], bF[4];
    #pragma unroll
    for (int mi = 0; mi < 2; ++mi)
      aF[mi] = __builtin_bit_cast(v16bf, *(const v16us*)&lA[cur][(waveM * 2 + mi) * 512 + l * 16]);
    #pragma unroll
    for (int ni = 0; ni < 4; ++ni)
      bF[ni] = __builtin_bit_cast(v16bf, *(const v16us*)&lB[cur][(waveN * 4 + ni) * 512 + l * 16]);

    #pragma unroll
    for (int mi = 0; mi < 2; ++mi)
      #pragma unroll
      for (int ni = 0; ni < 4; ++ni)
        acc[mi][ni] = __builtin_amdgcn_wmma_f32_16x16x32_bf16(
            false, aF[mi], false, bF[ni], (short)0, acc[mi][ni], false, false);
    __syncthreads();
  }

  // Epilogue (C/D layout: VGPR r -> M = r + 8*hi, N = l&15)
  const int rowBase = mtBase << 4, colBase = ntBase << 4;
  #pragma unroll
  for (int mi = 0; mi < 2; ++mi)
    #pragma unroll
    for (int ni = 0; ni < 4; ++ni) {
      const int col = colBase + waveN * 64 + ni * 16 + ln;
      const float bv = bias[col];
      #pragma unroll
      for (int r = 0; r < 8; ++r) {
        const int row = rowBase + waveM * 32 + mi * 16 + r + 8 * hi;
        float v = acc[mi][ni][r] + bv;
        if (EPI == 1) v = geluf(v);
        if (EPI == 2) v = sigf(v);
        if (EPI == 3) v += R[(size_t)row * N + col];
        C[(size_t)row * N + col] = v;
      }
    }
}

// ---------------------------------------------------------------------------
// Flash-attention on pre-tiled bf16 Q/K^T/V (per (b,h) slices of S*HDIM elems).
// Grid (B*H, SEQL/64), 128 thr = 4 waves; each wave owns 16 query rows.
__global__ __launch_bounds__(128)
void k_attn(const unsigned short* __restrict__ qT, const unsigned short* __restrict__ kT,
            const unsigned short* __restrict__ vT, const float* __restrict__ bw,
            float* __restrict__ O)
{
  __shared__ __attribute__((aligned(32))) unsigned short lP[4][16 * 32];
  const int bh = blockIdx.x, b = bh >> 4, h = bh & 15;
  const int wave = threadIdx.x >> 5;
  const int l = threadIdx.x & 31, ln = l & 15, hi = l >> 4;
  const int qt0 = blockIdx.y * 4 + wave;                 // q-tile 0..31
  const size_t sl = (size_t)bh * (SEQL * HDIM);
  const unsigned short* qs = qT + sl;
  const unsigned short* ks = kT + sl;  // B-tiled: K-dim=64 (2 kt), N-dim=512 keys (32 nt)
  const unsigned short* vs = vT + sl;  // B-tiled: K-dim=512 keys (16 kt), N-dim=64 (4 nt)
  const float scale = bw[h] * 0.125f;  // bw / sqrt(64)

  v16bf qF[2];
  #pragma unroll
  for (int ds = 0; ds < 2; ++ds)
    qF[ds] = __builtin_bit_cast(v16bf, *(const v16us*)&qs[(qt0 * 2 + ds) * 512 + l * 16]);

  const v8f vz = {0.f, 0.f, 0.f, 0.f, 0.f, 0.f, 0.f, 0.f};
  float m_i[8], l_i[8];
  v8f o[4];
  #pragma unroll
  for (int r = 0; r < 8; ++r) { m_i[r] = -1e30f; l_i[r] = 0.f; }
  #pragma unroll
  for (int ni = 0; ni < 4; ++ni) o[ni] = vz;

  for (int kc = 0; kc < SEQL; kc += 32) {
    v8f s0 = vz, s1 = vz;
    #pragma unroll
    for (int ds = 0; ds < 2; ++ds) {
      const v16bf b0 = __builtin_bit_cast(v16bf,
          *(const v16us*)&ks[(ds * 32 + (kc >> 4)) * 512 + l * 16]);
      const v16bf b1 = __builtin_bit_cast(v16bf,
          *(const v16us*)&ks[(ds * 32 + (kc >> 4) + 1) * 512 + l * 16]);
      s0 = __builtin_amdgcn_wmma_f32_16x16x32_bf16(false, qF[ds], false, b0, (short)0, s0, false, false);
      s1 = __builtin_amdgcn_wmma_f32_16x16x32_bf16(false, qF[ds], false, b1, (short)0, s1, false, false);
    }
    float corr[8];
    #pragma unroll
    for (int r = 0; r < 8; ++r) {
      s0[r] *= scale; s1[r] *= scale;
      float mx = fmaxf(s0[r], s1[r]);            // row r+8*hi lives in a 16-lane half
      mx = fmaxf(mx, __shfl_xor(mx, 1, 32));
      mx = fmaxf(mx, __shfl_xor(mx, 2, 32));
      mx = fmaxf(mx, __shfl_xor(mx, 4, 32));
      mx = fmaxf(mx, __shfl_xor(mx, 8, 32));
      const float mnew = fmaxf(m_i[r], mx);
      corr[r] = expf(m_i[r] - mnew);
      s0[r] = expf(s0[r] - mnew);
      s1[r] = expf(s1[r] - mnew);
      float rs = s0[r] + s1[r];
      rs += __shfl_xor(rs, 1, 32);
      rs += __shfl_xor(rs, 2, 32);
      rs += __shfl_xor(rs, 4, 32);
      rs += __shfl_xor(rs, 8, 32);
      l_i[r] = l_i[r] * corr[r] + rs;
      m_i[r] = mnew;
    }
    #pragma unroll
    for (int ni = 0; ni < 4; ++ni)
      #pragma unroll
      for (int r = 0; r < 8; ++r) o[ni][r] *= corr[r];

    __syncthreads();
    #pragma unroll
    for (int r = 0; r < 8; ++r) { // C-layout -> plain [m][j] in LDS
      lP[wave][(r + 8 * hi) * 32 + ln]      = f2bf(s0[r]);
      lP[wave][(r + 8 * hi) * 32 + 16 + ln] = f2bf(s1[r]);
    }
    __syncthreads();
    v16us up; // A-fragment of P (16x32)
    #pragma unroll
    for (int e = 0; e < 16; ++e) {
      const int j = (e < 8) ? (8 * hi + e) : (16 + 8 * hi + (e - 8));
      up[e] = lP[wave][ln * 32 + j];
    }
    const v16bf pF = __builtin_bit_cast(v16bf, up);
    #pragma unroll
    for (int ni = 0; ni < 4; ++ni) {
      const v16bf vF = __builtin_bit_cast(v16bf,
          *(const v16us*)&vs[((kc >> 5) * 4 + ni) * 512 + l * 16]);
      o[ni] = __builtin_amdgcn_wmma_f32_16x16x32_bf16(false, pF, false, vF, (short)0, o[ni], false, false);
    }
  }
  const size_t obase = (size_t)(b * SEQL) * DMODEL + h * HDIM;
  #pragma unroll
  for (int ni = 0; ni < 4; ++ni)
    #pragma unroll
    for (int r = 0; r < 8; ++r) {
      const int row = qt0 * 16 + r + 8 * hi;
      O[obase + (size_t)row * DMODEL + ni * 16 + ln] = o[ni][r] / l_i[r];
    }
}

// ---------------------------------------------------------------------------
// LayerNorm over D=1024, one block per token.
__global__ __launch_bounds__(256)
void k_ln(const float* __restrict__ X, const float* __restrict__ g,
          const float* __restrict__ b, float* __restrict__ Y)
{
  __shared__ float red[256];
  const int t = blockIdx.x, tid = threadIdx.x;
  const float* row = X + (size_t)t * DMODEL;
  float s = 0.f;
  for (int i = tid; i < DMODEL; i += 256) s += row[i];
  red[tid] = s; __syncthreads();
  for (int off = 128; off > 0; off >>= 1) { if (tid < off) red[tid] += red[tid + off]; __syncthreads(); }
  const float mu = red[0] * (1.0f / DMODEL);
  __syncthreads();
  float v = 0.f;
  for (int i = tid; i < DMODEL; i += 256) { const float d = row[i] - mu; v += d * d; }
  red[tid] = v; __syncthreads();
  for (int off = 128; off > 0; off >>= 1) { if (tid < off) red[tid] += red[tid + off]; __syncthreads(); }
  const float rstd = rsqrtf(red[0] * (1.0f / DMODEL) + 1e-5f);
  for (int i = tid; i < DMODEL; i += 256)
    Y[(size_t)t * DMODEL + i] = (row[i] - mu) * rstd * g[i] + b[i];
}

// ---------------------------------------------------------------------------
// Small / elementwise kernels
__global__ void k_embed(const int* __restrict__ ids, const float* __restrict__ emb,
                        const float* __restrict__ pos, float* __restrict__ x)
{
  const int idx = blockIdx.x * 256 + threadIdx.x;      // MTOK*DMODEL threads
  const int tok = idx / DMODEL, d = idx % DMODEL, s = tok % SEQL;
  x[idx] = emb[(size_t)ids[tok] * DMODEL + d] + pos[(size_t)s * DMODEL + d];
}
__global__ void k_concat2(const float* __restrict__ x, const float* __restrict__ p,
                          float* __restrict__ c)
{
  const int idx = blockIdx.x * 256 + threadIdx.x;      // MTOK*2*DMODEL threads
  const int m = idx / (2 * DMODEL), col = idx % (2 * DMODEL);
  c[idx] = (col < DMODEL) ? x[(size_t)m * DMODEL + col]
                          : p[(size_t)m * DMODEL + (col - DMODEL)];
}
__global__ void k_mix(float* x, const float* __restrict__ p, const float* __restrict__ g)
{
  const int idx = blockIdx.x * 256 + threadIdx.x;      // MTOK*DMODEL
  const float gv = g[idx];
  x[idx] = x[idx] * gv + p[idx] * (1.0f - gv);
}
__global__ void k_copy(float* dst, const float* __restrict__ src)
{
  const int idx = blockIdx.x * 256 + threadIdx.x;
  dst[idx] = src[idx];
}
__global__ void k_pool(const float* __restrict__ x, float* __restrict__ pooled)
{
  const int idx = blockIdx.x * 256 + threadIdx.x;      // BATCH*DMODEL
  const int b = idx / DMODEL, d = idx % DMODEL;
  float s = 0.f;
  for (int t = 0; t < SEQL; ++t) s += x[(size_t)(b * SEQL + t) * DMODEL + d];
  pooled[idx] = s * (1.0f / SEQL);
}
// tiny GEMM: C[m,n] = act(A@W + bias) ; act 0=none 1=gelu 2=tanh 3=sigmoid
__global__ void k_small(const float* __restrict__ A, const float* __restrict__ W,
                        const float* __restrict__ Bv, float* __restrict__ C,
                        int M, int N, int K, int act)
{
  const int idx = blockIdx.x * 256 + threadIdx.x;
  if (idx >= M * N) return;
  const int m = idx / N, n = idx % N;
  float s = Bv[n];
  for (int k = 0; k < K; ++k) s += A[(size_t)m * K + k] * W[(size_t)k * N + n];
  if (act == 1) s = geluf(s); else if (act == 2) s = tanhf(s); else if (act == 3) s = sigf(s);
  C[idx] = s;
}
__global__ void k_agg(const float* __restrict__ hb, float* __restrict__ agg)
{
  const int j = threadIdx.x;
  if (j < NBASIN) {
    float s = 0.f;
    for (int b = 0; b < BATCH; ++b) s += hb[b * NBASIN + j];
    agg[j] = s * (1.0f / BATCH);
  }
}
__global__ void k_cat64(const float* __restrict__ a, const float* __restrict__ b,
                        float* __restrict__ c)
{
  const int i = threadIdx.x;
  if (i < NBASIN) { c[i] = a[i]; c[NBASIN + i] = b[i]; }
}
__global__ void k_bupdate(float* basin, const float* __restrict__ agg,
                          const float* __restrict__ g)
{
  const int i = threadIdx.x;
  if (i < NBASIN) { const float gv = g[i]; basin[i] = basin[i] * (1.0f - gv) + agg[i] * gv; }
}
__global__ void k_copy64(float* dst, const float* __restrict__ src)
{
  const int i = threadIdx.x;
  if (i < NBASIN) dst[i] = src[i];
}

// ---------------------------------------------------------------------------
extern "C" void kernel_launch(void* const* d_in, const int* in_sizes, int n_in,
                              void* d_out, int out_size, void* d_ws, size_t ws_size,
                              hipStream_t stream)
{
  (void)in_sizes; (void)n_in; (void)out_size; (void)ws_size;
  // Input order: JAX pytree flatten = dict keys sorted alphabetically.
  const int*   ids    = (const int*)d_in[0];
  const float* basin0 = (const float*)d_in[1];
  const float* bu_b   = (const float*)d_in[2];
  const float* bu_w   = (const float*)d_in[3];
  const float* emb    = (const float*)d_in[4];
  const float* enc_b1 = (const float*)d_in[5];
  const float* enc_b2 = (const float*)d_in[6];
  const float* enc_w1 = (const float*)d_in[7];
  const float* enc_w2 = (const float*)d_in[8];
  const float* out_b  = (const float*)d_in[89];
  const float* out_w  = (const float*)d_in[90];
  const float* pos    = (const float*)d_in[91];

  struct Lp { const float *b1,*b2,*bg,*bk,*bm,*bo,*bq,*bv,
                          *ln1_b,*ln1_g,*ln2_b,*ln2_g,
                          *w1,*w2,*wg,*wk,*wm,*wo,*wq,*wv; };
  Lp L[NLAYERS];
  for (int li = 0; li < NLAYERS; ++li) {
    const int o = 9 + li * 20; // sorted keys within each layer dict
    L[li].b1    = (const float*)d_in[o+0];  L[li].b2    = (const float*)d_in[o+1];
    L[li].bg    = (const float*)d_in[o+2];  L[li].bk    = (const float*)d_in[o+3];
    L[li].bm    = (const float*)d_in[o+4];  L[li].bo    = (const float*)d_in[o+5];
    L[li].bq    = (const float*)d_in[o+6];  L[li].bv    = (const float*)d_in[o+7];
    L[li].ln1_b = (const float*)d_in[o+8];  L[li].ln1_g = (const float*)d_in[o+9];
    L[li].ln2_b = (const float*)d_in[o+10]; L[li].ln2_g = (const float*)d_in[o+11];
    L[li].w1    = (const float*)d_in[o+12]; L[li].w2    = (const float*)d_in[o+13];
    L[li].wg    = (const float*)d_in[o+14]; L[li].wk    = (const float*)d_in[o+15];
    L[li].wm    = (const float*)d_in[o+16]; L[li].wo    = (const float*)d_in[o+17];
    L[li].wq    = (const float*)d_in[o+18]; L[li].wv    = (const float*)d_in[o+19];
  }

  // ---- workspace carving ----
  char* cur = (char*)d_ws;
  auto alloc_f = [&](size_t n) -> float* {
    float* p = (float*)cur; cur += ((n * 4 + 63) & ~(size_t)63); return p; };
  auto alloc_h = [&](size_t n) -> unsigned short* {
    unsigned short* p = (unsigned short*)cur; cur += ((n * 2 + 63) & ~(size_t)63); return p; };

  const size_t TD = (size_t)MTOK * DMODEL;
  float* x      = alloc_f(TD);
  float* xn     = alloc_f(TD);
  float* tf     = alloc_f(TD);                  // q/k/v f32 temp (reused)
  float* ao     = alloc_f(TD);
  float* hbuf   = alloc_f((size_t)MTOK * FFDIM);
  float* comb   = alloc_f((size_t)MTOK * 2 * DMODEL);
  float* gbuf   = alloc_f(TD);
  float* prev   = alloc_f((size_t)NLAYERS * TD);
  float* pooled = alloc_f(BATCH * DMODEL);
  float* ench   = alloc_f(BATCH * (DMODEL / 2));
  float* hb     = alloc_f(BATCH * NBASIN);
  float* agg    = alloc_f(NBASIN);
  float* basin  = alloc_f(NBASIN);
  float* cbuf   = alloc_f(2 * NBASIN);
  float* bwv    = alloc_f(NHEADS);
  float* bgate  = alloc_f(NBASIN);

  unsigned short* xnT   = alloc_h(TD);
  unsigned short* qTb   = alloc_h(TD);
  unsigned short* kTb   = alloc_h(TD);
  unsigned short* vTb   = alloc_h(TD);
  unsigned short* aoT   = alloc_h(TD);
  unsigned short* hT    = alloc_h((size_t)MTOK * FFDIM);
  unsigned short* combT = alloc_h((size_t)MTOK * 2 * DMODEL);
  unsigned short* xT    = alloc_h(TD);
  unsigned short* wcur  = alloc_h(91488256);    // weight arena (all B-tiled)

  // ---- one-time weight tiling (B-fragment order) ----
  auto tileW = [&](const float* w, int K, int N) -> const unsigned short* {
    unsigned short* dst = wcur; wcur += (size_t)K * N;
    const int blocks = (int)(((size_t)K * N / 16) / 256);
    k_cvtB<<<blocks, 256, 0, stream>>>(w, dst, K, N, N, 1, 1, 0, 0);
    return dst;
  };
  const unsigned short *WQ[NLAYERS], *WK[NLAYERS], *WV[NLAYERS], *WO[NLAYERS],
                       *W1[NLAYERS], *W2[NLAYERS], *WG[NLAYERS];
  for (int li = 0; li < NLAYERS; ++li) {
    WQ[li] = tileW(L[li].wq, DMODEL, DMODEL);
    WK[li] = tileW(L[li].wk, DMODEL, DMODEL);
    WV[li] = tileW(L[li].wv, DMODEL, DMODEL);
    WO[li] = tileW(L[li].wo, DMODEL, DMODEL);
    W1[li] = tileW(L[li].w1, DMODEL, FFDIM);
    W2[li] = tileW(L[li].w2, FFDIM, DMODEL);
    WG[li] = tileW(L[li].wg, 2 * DMODEL, DMODEL);
  }
  const unsigned short* OW = tileW(out_w, DMODEL, VOCAB);

  auto G = [&](int epi, const unsigned short* A, const unsigned short* B,
               const float* bias, const float* R, float* C, int M, int N, int K) {
    dim3 g(N / 128, M / 128), blk(256);
    switch (epi) {
      case 0: k_gemm<0><<<g, blk, 0, stream>>>(A, B, bias, R, C, M, N, K); break;
      case 1: k_gemm<1><<<g, blk, 0, stream>>>(A, B, bias, R, C, M, N, K); break;
      case 2: k_gemm<2><<<g, blk, 0, stream>>>(A, B, bias, R, C, M, N, K); break;
      default: k_gemm<3><<<g, blk, 0, stream>>>(A, B, bias, R, C, M, N, K); break;
    }
  };
  auto cvtA_plain = [&](const float* s, unsigned short* d, int M, int K) {
    k_cvtA<<<(int)(((size_t)M * K / 16) / 256), 256, 0, stream>>>(s, d, M, K, K, 1, 0, 0);
  };

  const int gTD  = (int)(TD / 256);
  const int gTD2 = (int)((size_t)MTOK * 2 * DMODEL / 256);
  const long SD  = (long)SEQL * DMODEL;

  k_copy64<<<1, 64, 0, stream>>>(basin, basin0);
  k_embed<<<gTD, 256, 0, stream>>>(ids, emb, pos, x);

  for (int pass = 0; pass < 2; ++pass) {
    for (int li = 0; li < NLAYERS; ++li) {
      const Lp& lp = L[li];
      float* pv = prev + (size_t)li * TD;
      if (pass > 0) {
        k_concat2<<<gTD2, 256, 0, stream>>>(x, pv, comb);
        cvtA_plain(comb, combT, MTOK, 2 * DMODEL);
        G(2, combT, WG[li], lp.bg, nullptr, gbuf, MTOK, DMODEL, 2 * DMODEL);
        k_mix<<<gTD, 256, 0, stream>>>(x, pv, gbuf);
      }
      k_ln<<<MTOK, 256, 0, stream>>>(x, lp.ln1_g, lp.ln1_b, xn);
      cvtA_plain(xn, xnT, MTOK, DMODEL);
      // Q: A-tiled per (b,h) slice
      G(0, xnT, WQ[li], lp.bq, nullptr, tf, MTOK, DMODEL, DMODEL);
      k_cvtA<<<1024, 256, 0, stream>>>(tf, qTb, SEQL, HDIM, DMODEL, NHEADS, (long)HDIM, SD);
      // K^T: B-tiled (k-dim = head dim, n-dim = keys)
      G(0, xnT, WK[li], lp.bk, nullptr, tf, MTOK, DMODEL, DMODEL);
      k_cvtB<<<1024, 256, 0, stream>>>(tf, kTb, HDIM, SEQL, 1, DMODEL, NHEADS, (long)HDIM, SD);
      // V: B-tiled (k-dim = keys, n-dim = head dim)
      G(0, xnT, WV[li], lp.bv, nullptr, tf, MTOK, DMODEL, DMODEL);
      k_cvtB<<<1024, 256, 0, stream>>>(tf, vTb, SEQL, HDIM, DMODEL, 1, NHEADS, (long)HDIM, SD);
      k_small<<<1, 256, 0, stream>>>(basin, lp.wm, lp.bm, bwv, 1, NHEADS, NBASIN, 3);
      k_attn<<<dim3(BATCH * NHEADS, SEQL / 64), 128, 0, stream>>>(qTb, kTb, vTb, bwv, ao);
      cvtA_plain(ao, aoT, MTOK, DMODEL);
      G(3, aoT, WO[li], lp.bo, x, x, MTOK, DMODEL, DMODEL);          // x += attn@wo+bo
      k_ln<<<MTOK, 256, 0, stream>>>(x, lp.ln2_g, lp.ln2_b, xn);
      cvtA_plain(xn, xnT, MTOK, DMODEL);
      G(1, xnT, W1[li], lp.b1, nullptr, hbuf, MTOK, FFDIM, DMODEL);  // gelu
      cvtA_plain(hbuf, hT, MTOK, FFDIM);
      G(3, hT, W2[li], lp.b2, x, x, MTOK, DMODEL, FFDIM);            // x += h@w2+b2
      k_copy<<<gTD, 256, 0, stream>>>(pv, x);                         // layer state
    }
    if (pass == 0) { // basin encode + update (unused after last pass)
      k_pool<<<BATCH * DMODEL / 256, 256, 0, stream>>>(x, pooled);
      k_small<<<(BATCH * (DMODEL/2) + 255) / 256, 256, 0, stream>>>(
          pooled, enc_w1, enc_b1, ench, BATCH, DMODEL / 2, DMODEL, 1);
      k_small<<<(BATCH * NBASIN + 255) / 256, 256, 0, stream>>>(
          ench, enc_w2, enc_b2, hb, BATCH, NBASIN, DMODEL / 2, 2);
      k_agg<<<1, 64, 0, stream>>>(hb, agg);
      k_cat64<<<1, 64, 0, stream>>>(basin, agg, cbuf);
      k_small<<<1, 256, 0, stream>>>(cbuf, bu_w, bu_b, bgate, 1, NBASIN, 2 * NBASIN, 3);
      k_bupdate<<<1, 64, 0, stream>>>(basin, agg, bgate);
    }
  }
  // logits: [4096,1024] @ [1024,32000] + out_b -> d_out (f32)
  cvtA_plain(x, xT, MTOK, DMODEL);
  G(0, xT, OW, out_b, nullptr, (float*)d_out, MTOK, VOCAB, DMODEL);
}
